// GAT_Modelv4_38826504356233
// MI455X (gfx1250) — matrix operand order repaired and checked by reading, not compiled, not run on hardware
//
#include <hip/hip_runtime.h>
#include <math.h>
#include <string.h>

// ---------------------------------------------------------------------------
// GATv2 GNN forward for MI455X (gfx1250).
// Dense node-feature GEMMs run through v_wmma_f32_16x16x32_bf16 (f32 accum),
// register-blocked 32x32 per wave (2x2 WMMA tiles) to halve loads per WMMA.
// Edge softmax-aggregation uses hw float atomics (global_atomic_add_f32 /
// global_atomic_max_u32 with monotone float encoding).
// ---------------------------------------------------------------------------

#define N_NODES   100000
#define N_EDGES   800000
#define ET_EDGES  900000      // edges + self loops
#define N_HEADS   3
#define N_GRAPHS  256

typedef unsigned int u32;
typedef unsigned short u16;

typedef __attribute__((ext_vector_type(16))) __bf16 v16bf;
typedef __attribute__((ext_vector_type(8)))  float  v8f;

// ---------------- helpers ----------------
__device__ __forceinline__ u16 f2bf(float f) {
  u32 u = __float_as_uint(f);
  u32 r = 0x7FFFu + ((u >> 16) & 1u);        // round-to-nearest-even
  return (u16)((u + r) >> 16);
}
// monotone float -> uint mapping so unsigned atomicMax == float max
__device__ __forceinline__ u32 encF(float x) {
  u32 u = __float_as_uint(x);
  return (u & 0x80000000u) ? ~u : (u | 0x80000000u);
}
__device__ __forceinline__ float decF(u32 u) {
  return (u & 0x80000000u) ? __uint_as_float(u & 0x7FFFFFFFu)
                           : __uint_as_float(~u);
}

// ---------------- conversion / fill kernels ----------------
__global__ void k_cvt_pad(const float* __restrict__ src, u16* __restrict__ dst,
                          int K, int Kpad) {
  size_t i = (size_t)blockIdx.x * blockDim.x + threadIdx.x;
  size_t total = (size_t)N_NODES * Kpad;
  if (i >= total) return;
  int k = (int)(i % Kpad);
  size_t r = i / Kpad;
  dst[i] = (k < K) ? f2bf(src[r * (size_t)K + k]) : (u16)0;
}

// W [K, Dout] f32 -> WT [Dout, Kpad] bf16 (zero-padded K)
__global__ void k_cvtW_T(const float* __restrict__ W, u16* __restrict__ WT,
                         int K, int Dout, int Kpad) {
  int i = blockIdx.x * blockDim.x + threadIdx.x;
  if (i >= Dout * Kpad) return;
  int k = i % Kpad;
  int o = i / Kpad;
  WT[i] = (k < K) ? f2bf(W[(size_t)k * Dout + o]) : (u16)0;
}

__global__ void k_fill_f32(float* p, float v, size_t n) {
  size_t i = (size_t)blockIdx.x * blockDim.x + threadIdx.x;
  if (i < n) p[i] = v;
}
__global__ void k_fill_u32(u32* p, u32 v, size_t n) {
  size_t i = (size_t)blockIdx.x * blockDim.x + threadIdx.x;
  if (i < n) p[i] = v;
}

// decode encoded-max buffer in place, non-finite -> 0 (reference semantics)
__global__ void k_decode(u32* __restrict__ p, size_t n) {
  size_t i = (size_t)blockIdx.x * blockDim.x + threadIdx.x;
  if (i >= n) return;
  float f = decF(p[i]);
  if (!isfinite(f)) f = 0.f;
  ((float*)p)[i] = f;
}

// ---------------- WMMA GEMM: Y[N,Dout] = X[N,Kpad]@W + bias ----------------
// X bf16 row-major (stride Kpad), WT bf16 = W^T row-major [Dout, Kpad].
// One wave computes a 32x32 output block = 2x2 WMMA tiles; K-loop step 32.
// A fragment layout (16x32 bf16): lanes 0-15 hold row M=lane, K {0..7,16..23};
// lanes 16-31 hold row M=lane-16, K {8..15,24..31}.
// B fragment layout (32x16 bf16): lanes 0-15 hold col N=lane, K {0..7,16..23};
// lanes 16-31 hold col N=lane-16, K {8..15,24..31}.
__global__ void k_gemm_bf16(const u16* __restrict__ X, const u16* __restrict__ WT,
                            const float* __restrict__ bias, float* __restrict__ Y,
                            int Mt, int Nt, int Kpad, int Dout) {
  int gtid = blockIdx.x * blockDim.x + threadIdx.x;
  int wave = gtid >> 5;
  int lane = gtid & 31;
  if (wave >= Mt * Nt) return;   // wave-uniform; EXEC stays all-ones
  int mt = wave / Nt;
  int nt = wave - mt * Nt;
  int row0 = mt * 32 + (lane & 15);
  int col0 = nt * 32 + (lane & 15);
  int kA = (lane < 16) ? 0 : 8;
  int kB = (lane < 16) ? 0 : 8;
  const u16* ap0 = X  + (size_t)row0 * Kpad + kA;
  const u16* ap1 = ap0 + (size_t)16 * Kpad;
  const u16* bp0 = WT + (size_t)col0 * Kpad + kB;
  const u16* bp1 = bp0 + (size_t)16 * Kpad;
  v8f acc00 = {}, acc01 = {}, acc10 = {}, acc11 = {};
  union U { uint4 q[2]; v16bf v; };
  for (int k0 = 0; k0 < Kpad; k0 += 32) {
    U a0, a1, b0, b1;
    a0.q[0] = *(const uint4*)(ap0 + k0);
    a0.q[1] = *(const uint4*)(ap0 + k0 + 16);
    a1.q[0] = *(const uint4*)(ap1 + k0);
    a1.q[1] = *(const uint4*)(ap1 + k0 + 16);
    b0.q[0] = *(const uint4*)(bp0 + k0);
    b0.q[1] = *(const uint4*)(bp0 + k0 + 16);
    b1.q[0] = *(const uint4*)(bp1 + k0);
    b1.q[1] = *(const uint4*)(bp1 + k0 + 16);
    acc00 = __builtin_amdgcn_wmma_f32_16x16x32_bf16(false, a0.v, false, b0.v, (short)0, acc00, false, false);
    acc01 = __builtin_amdgcn_wmma_f32_16x16x32_bf16(false, a0.v, false, b1.v, (short)0, acc01, false, false);
    acc10 = __builtin_amdgcn_wmma_f32_16x16x32_bf16(false, a1.v, false, b0.v, (short)0, acc10, false, false);
    acc11 = __builtin_amdgcn_wmma_f32_16x16x32_bf16(false, a1.v, false, b1.v, (short)0, acc11, false, false);
  }
  // C/D layout: lane holds col N=lane&15; VGPR r -> row (lane<16?0:8)+r
  float bv0 = bias[col0];
  float bv1 = bias[col0 + 16];
  int mb = mt * 32 + ((lane < 16) ? 0 : 8);
  float* y00 = Y + (size_t)mb * Dout + col0;        // rows mb..mb+7,  cols col0
  float* y01 = y00 + 16;                            // cols col0+16
  float* y10 = y00 + (size_t)16 * Dout;             // rows +16
  float* y11 = y10 + 16;
#pragma unroll
  for (int r = 0; r < 8; ++r) {
    y00[(size_t)r * Dout] = acc00[r] + bv0;
    y01[(size_t)r * Dout] = acc01[r] + bv1;
    y10[(size_t)r * Dout] = acc10[r] + bv0;
    y11[(size_t)r * Dout] = acc11[r] + bv1;
  }
}

// ---------------- edge kernels ----------------
__global__ void k_edge_logit(const float* __restrict__ XL, const float* __restrict__ XR,
                             const float* __restrict__ att, const int* __restrict__ ei,
                             float* __restrict__ logit, u32* __restrict__ lmaxu, int C) {
  int idx = blockIdx.x * blockDim.x + threadIdx.x;
  if (idx >= ET_EDGES * N_HEADS) return;
  int e = idx / N_HEADS;
  int h = idx - e * N_HEADS;
  int s, d;
  if (e < N_EDGES) { s = ei[e]; d = ei[N_EDGES + e]; } else { s = e - N_EDGES; d = s; }
  const float4* pl = (const float4*)(XL + ((size_t)s * N_HEADS + h) * C);
  const float4* pr = (const float4*)(XR + ((size_t)d * N_HEADS + h) * C);
  const float4* pa = (const float4*)(att + (size_t)h * C);
  float acc = 0.f;
  int c4 = C >> 2;
  for (int c = 0; c < c4; ++c) {
    float4 a = pl[c], b = pr[c], w = pa[c];
    float m;
    m = a.x + b.x; m = (m > 0.f) ? m : 0.2f * m; acc += m * w.x;
    m = a.y + b.y; m = (m > 0.f) ? m : 0.2f * m; acc += m * w.y;
    m = a.z + b.z; m = (m > 0.f) ? m : 0.2f * m; acc += m * w.z;
    m = a.w + b.w; m = (m > 0.f) ? m : 0.2f * m; acc += m * w.w;
  }
  logit[idx] = acc;
  atomicMax(&lmaxu[(size_t)d * N_HEADS + h], encF(acc));
}

__global__ void k_edge_accum(const float* __restrict__ XL, const float* __restrict__ logit,
                             const float* __restrict__ lmax, const int* __restrict__ ei,
                             float* __restrict__ ACC, float* __restrict__ DEN, int C) {
  int idx = blockIdx.x * blockDim.x + threadIdx.x;
  if (idx >= ET_EDGES * N_HEADS) return;
  int e = idx / N_HEADS;
  int h = idx - e * N_HEADS;
  int s, d;
  if (e < N_EDGES) { s = ei[e]; d = ei[N_EDGES + e]; } else { s = e - N_EDGES; d = s; }
  float ex = expf(logit[idx] - lmax[(size_t)d * N_HEADS + h]);
  unsafeAtomicAdd(&DEN[(size_t)d * N_HEADS + h], ex);
  const float* pl = XL + ((size_t)s * N_HEADS + h) * C;
  float* pa = ACC + ((size_t)d * N_HEADS + h) * C;
  for (int c = 0; c < C; c += 4) {
    float4 a = *(const float4*)(pl + c);
    unsafeAtomicAdd(pa + c + 0, ex * a.x);
    unsafeAtomicAdd(pa + c + 1, ex * a.y);
    unsafeAtomicAdd(pa + c + 2, ex * a.z);
    unsafeAtomicAdd(pa + c + 3, ex * a.w);
  }
}

// out = relu(BN(acc/den + gat_bias)) [+ residual], concat layout [N, H*C]
__global__ void k_finalize_concat(const float* __restrict__ ACC, const float* __restrict__ DEN,
                                  const float* __restrict__ gb, const float* __restrict__ gam,
                                  const float* __restrict__ bet, const float* __restrict__ mu,
                                  const float* __restrict__ var, const float* __restrict__ res,
                                  float* __restrict__ OUT, int HC, int C) {
  size_t i = (size_t)blockIdx.x * blockDim.x + threadIdx.x;
  size_t total = (size_t)N_NODES * HC;
  if (i >= total) return;
  int j = (int)(i % HC);
  size_t n = i / HC;
  int h = j / C;
  float v = ACC[i] / (DEN[n * N_HEADS + h] + 1e-16f) + gb[j];
  v = (v - mu[j]) * (gam[j] * rsqrtf(var[j] + 1e-5f)) + bet[j];
  v = fmaxf(v, 0.f);
  if (res) v += res[i];
  OUT[i] = v;
}

// concat=False: mean over heads, then +bias, BN, relu
__global__ void k_finalize_mean(const float* __restrict__ ACC, const float* __restrict__ DEN,
                                const float* __restrict__ gb, const float* __restrict__ gam,
                                const float* __restrict__ bet, const float* __restrict__ mu,
                                const float* __restrict__ var, float* __restrict__ OUT, int C) {
  size_t i = (size_t)blockIdx.x * blockDim.x + threadIdx.x;
  size_t total = (size_t)N_NODES * C;
  if (i >= total) return;
  int c = (int)(i % C);
  size_t n = i / C;
  float s = 0.f;
#pragma unroll
  for (int h = 0; h < N_HEADS; ++h)
    s += ACC[((size_t)n * N_HEADS + h) * C + c] / (DEN[n * N_HEADS + h] + 1e-16f);
  float v = s * (1.f / 3.f) + gb[c];
  v = (v - mu[c]) * (gam[c] * rsqrtf(var[c] + 1e-5f)) + bet[c];
  OUT[i] = fmaxf(v, 0.f);
}

// ---------------- readout kernels ----------------
__global__ void k_readout1(const float* __restrict__ H5, const int* __restrict__ batch,
                           const float* __restrict__ tp, float* __restrict__ CNT,
                           float* __restrict__ SUM, float* __restrict__ SUM2,
                           u32* __restrict__ MXU, u32* __restrict__ GMAXU) {
  int i = blockIdx.x * blockDim.x + threadIdx.x;
  if (i >= N_NODES * 128) return;
  int c = i & 127;
  int n = i >> 7;
  int g = batch[n];
  float v = H5[i];
  unsafeAtomicAdd(&SUM[g * 128 + c], v);
  unsafeAtomicAdd(&SUM2[g * 128 + c], v * v);
  atomicMax(&MXU[g * 128 + c], encF(v));
  atomicMax(&GMAXU[g * 128 + c], encF(v * tp[0]));
  if (c == 0) unsafeAtomicAdd(&CNT[g], 1.0f);
}

__global__ void k_readout2(const float* __restrict__ H5, const int* __restrict__ batch,
                           const float* __restrict__ tp, const float* __restrict__ GMAX,
                           float* __restrict__ RDEN, float* __restrict__ SMN) {
  int i = blockIdx.x * blockDim.x + threadIdx.x;
  if (i >= N_NODES * 128) return;
  int c = i & 127;
  int n = i >> 7;
  int g = batch[n];
  float v = H5[i];
  float ex = expf(v * tp[0] - GMAX[g * 128 + c]);
  unsafeAtomicAdd(&RDEN[g * 128 + c], ex);
  unsafeAtomicAdd(&SMN[g * 128 + c], ex * v);
}

__global__ void k_readout3(const float* __restrict__ CNT, const float* __restrict__ SUM,
                           const float* __restrict__ SUM2, const float* __restrict__ MX,
                           const float* __restrict__ SMN, const float* __restrict__ RDEN,
                           float* __restrict__ R) {
  int i = blockIdx.x * blockDim.x + threadIdx.x;
  if (i >= N_GRAPHS * 128) return;
  int c = i & 127;
  int g = i >> 7;
  float cnt = fmaxf(CNT[g], 1.f);
  float mean = SUM[i] / cnt;
  float m2 = SUM2[i] / cnt;
  float sd = sqrtf(fmaxf(m2 - mean * mean, 0.f) + 1e-5f);
  float sm = SMN[i] / (RDEN[i] + 1e-16f);
  float* r = R + (size_t)g * 512;
  r[c] = mean;
  r[128 + c] = sd;
  r[256 + c] = MX[i];
  r[384 + c] = sm;
}

// ---------------- small dense FC (no activation between layers) ----------------
__global__ void k_fc(const float* __restrict__ X, const float* __restrict__ W,
                     const float* __restrict__ b, float* __restrict__ Y,
                     int G, int K, int Dout) {
  int i = blockIdx.x * blockDim.x + threadIdx.x;
  if (i >= G * Dout) return;
  int o = i % Dout;
  int g = i / Dout;
  float s = b[o];
  const float* xp = X + (size_t)g * K;
  for (int k = 0; k < K; ++k) s += xp[k] * W[(size_t)k * Dout + o];
  Y[i] = s;
}

// ---------------------------------------------------------------------------
extern "C" void kernel_launch(void* const* d_in, const int* in_sizes, int n_in,
                              void* d_out, int out_size, void* d_ws, size_t ws_size,
                              hipStream_t stream) {
  (void)in_sizes; (void)n_in; (void)out_size; (void)ws_size;
  const float* x    = (const float*)d_in[0];
  const int* ei     = (const int*)d_in[1];
  const int* batch  = (const int*)d_in[2];
  // params flattened in dict order:
  // conv{1..5}: Wl,bl,Wr,br,att,bias ; bn{1..5}: gamma,beta,mean,var ; t ; fc*
  const float* tp   = (const float*)d_in[53];
  const float* fc1W = (const float*)d_in[54];
  const float* fc1b = (const float*)d_in[55];
  const float* fc2W = (const float*)d_in[56];
  const float* fc2b = (const float*)d_in[57];
  const float* fc3W = (const float*)d_in[58];
  const float* fc3b = (const float*)d_in[59];

  // ---- workspace carve (256B aligned regions) ----
  char* ws = (char*)d_ws;
  size_t off = 0;
  auto carve = [&](size_t bytes) -> char* {
    char* p = ws + off;
    off += (bytes + 255) & ~(size_t)255;
    return p;
  };
  u16*   XB    = (u16*)carve((size_t)N_NODES * 384 * 2);
  u16*   WTL   = (u16*)carve((size_t)384 * 384 * 2);
  u16*   WTR   = (u16*)carve((size_t)384 * 384 * 2);
  float* XLb   = (float*)carve((size_t)N_NODES * 384 * 4);
  float* XRb   = (float*)carve((size_t)N_NODES * 384 * 4);  // reused as ACC
  float* H2b   = (float*)carve((size_t)N_NODES * 384 * 4);
  float* HCUR  = (float*)carve((size_t)N_NODES * 384 * 4);
  float* LOGIT = (float*)carve((size_t)ET_EDGES * 3 * 4);
  u32*   LMAXU = (u32*)carve((size_t)N_NODES * 3 * 4);
  float* DEN   = (float*)carve((size_t)N_NODES * 3 * 4);
  float* CNT   = (float*)carve(N_GRAPHS * 4);
  float* SUM   = (float*)carve(N_GRAPHS * 128 * 4);
  float* SUM2  = (float*)carve(N_GRAPHS * 128 * 4);
  u32*   MXU   = (u32*)carve(N_GRAPHS * 128 * 4);
  u32*   GMAXU = (u32*)carve(N_GRAPHS * 128 * 4);
  float* RDEN  = (float*)carve(N_GRAPHS * 128 * 4);
  float* SMN   = (float*)carve(N_GRAPHS * 128 * 4);
  float* Rm    = (float*)carve(N_GRAPHS * 512 * 4);
  float* T1    = (float*)carve(N_GRAPHS * 128 * 4);
  float* T2    = (float*)carve(N_GRAPHS * 64 * 4);
  float* ACC   = XRb;             // free after k_edge_logit
  float* LMAXF = (float*)LMAXU;   // decoded in place
  float* H5    = XLb;             // free after last k_edge_accum

  // host-side encF(-1e30f)  (negative -> ~bits)
  u32 encNeg;
  { float f = -1e30f; u32 u; memcpy(&u, &f, 4); encNeg = ~u; }

  auto blocks = [](size_t n) -> unsigned { return (unsigned)((n + 255) / 256); };

  // ---- one GATv2 layer + BN + relu (+ residual) ----
  auto layer = [&](const float* in, int din, int C, bool concat,
                   const float* res, float* out, int pbase, int bnbase) {
    int HC = 3 * C;
    int Kpad = (din + 31) & ~31;
    const float* Wl  = (const float*)d_in[pbase + 0];
    const float* bl  = (const float*)d_in[pbase + 1];
    const float* Wr  = (const float*)d_in[pbase + 2];
    const float* br  = (const float*)d_in[pbase + 3];
    const float* att = (const float*)d_in[pbase + 4];
    const float* gb  = (const float*)d_in[pbase + 5];
    const float* gam = (const float*)d_in[bnbase + 0];
    const float* bet = (const float*)d_in[bnbase + 1];
    const float* mu  = (const float*)d_in[bnbase + 2];
    const float* var = (const float*)d_in[bnbase + 3];

    size_t ncv = (size_t)N_NODES * Kpad;
    k_cvt_pad<<<blocks(ncv), 256, 0, stream>>>(in, XB, din, Kpad);
    size_t nwv = (size_t)HC * Kpad;
    k_cvtW_T<<<blocks(nwv), 256, 0, stream>>>(Wl, WTL, din, HC, Kpad);
    k_cvtW_T<<<blocks(nwv), 256, 0, stream>>>(Wr, WTR, din, HC, Kpad);

    int Mt = N_NODES / 32, Nt = HC / 32;   // 32x32 block per wave
    size_t thr = (size_t)Mt * Nt * 32;
    k_gemm_bf16<<<blocks(thr), 256, 0, stream>>>(XB, WTL, bl, XLb, Mt, Nt, Kpad, HC);
    k_gemm_bf16<<<blocks(thr), 256, 0, stream>>>(XB, WTR, br, XRb, Mt, Nt, Kpad, HC);

    k_fill_u32<<<blocks((size_t)N_NODES * 3), 256, 0, stream>>>(LMAXU, encNeg, (size_t)N_NODES * 3);
    size_t eth = (size_t)ET_EDGES * 3;
    k_edge_logit<<<blocks(eth), 256, 0, stream>>>(XLb, XRb, att, ei, LOGIT, LMAXU, C);
    k_decode<<<blocks((size_t)N_NODES * 3), 256, 0, stream>>>(LMAXU, (size_t)N_NODES * 3);

    k_fill_f32<<<blocks((size_t)N_NODES * HC), 256, 0, stream>>>(ACC, 0.f, (size_t)N_NODES * HC);
    k_fill_f32<<<blocks((size_t)N_NODES * 3), 256, 0, stream>>>(DEN, 0.f, (size_t)N_NODES * 3);
    k_edge_accum<<<blocks(eth), 256, 0, stream>>>(XLb, LOGIT, LMAXF, ei, ACC, DEN, C);

    if (concat)
      k_finalize_concat<<<blocks((size_t)N_NODES * HC), 256, 0, stream>>>(
          ACC, DEN, gb, gam, bet, mu, var, res, out, HC, C);
    else
      k_finalize_mean<<<blocks((size_t)N_NODES * C), 256, 0, stream>>>(
          ACC, DEN, gb, gam, bet, mu, var, out, C);
  };

  // ---- network ----
  layer(x,    16,  64,  true,  nullptr, HCUR, 3,  9);   // conv1/bn1 -> [N,192]
  layer(HCUR, 192, 128, true,  nullptr, H2b,  13, 19);  // conv2/bn2 -> [N,384]
  layer(H2b,  384, 128, true,  nullptr, HCUR, 23, 29);  // conv3/bn3
  layer(HCUR, 384, 128, true,  H2b,     HCUR, 33, 39);  // conv4/bn4, h4 = h2 + relu(bn(.))
  layer(HCUR, 384, 128, false, nullptr, H5,   43, 49);  // conv5/bn5 -> [N,128]

  // ---- readout ----
  k_fill_f32<<<1, 256, 0, stream>>>(CNT, 0.f, N_GRAPHS);
  k_fill_f32<<<blocks(N_GRAPHS * 128), 256, 0, stream>>>(SUM,  0.f, N_GRAPHS * 128);
  k_fill_f32<<<blocks(N_GRAPHS * 128), 256, 0, stream>>>(SUM2, 0.f, N_GRAPHS * 128);
  k_fill_f32<<<blocks(N_GRAPHS * 128), 256, 0, stream>>>(RDEN, 0.f, N_GRAPHS * 128);
  k_fill_f32<<<blocks(N_GRAPHS * 128), 256, 0, stream>>>(SMN,  0.f, N_GRAPHS * 128);
  k_fill_u32<<<blocks(N_GRAPHS * 128), 256, 0, stream>>>(MXU,   encNeg, N_GRAPHS * 128);
  k_fill_u32<<<blocks(N_GRAPHS * 128), 256, 0, stream>>>(GMAXU, encNeg, N_GRAPHS * 128);

  k_readout1<<<blocks((size_t)N_NODES * 128), 256, 0, stream>>>(H5, batch, tp, CNT, SUM, SUM2, MXU, GMAXU);
  k_decode<<<blocks(N_GRAPHS * 128), 256, 0, stream>>>(MXU,   N_GRAPHS * 128);
  k_decode<<<blocks(N_GRAPHS * 128), 256, 0, stream>>>(GMAXU, N_GRAPHS * 128);
  k_readout2<<<blocks((size_t)N_NODES * 128), 256, 0, stream>>>(H5, batch, tp, (float*)GMAXU, RDEN, SMN);
  k_readout3<<<blocks(N_GRAPHS * 128), 256, 0, stream>>>(CNT, SUM, SUM2, (float*)MXU, SMN, RDEN, Rm);

  // ---- MLP head ----
  k_fc<<<blocks(N_GRAPHS * 128), 256, 0, stream>>>(Rm, fc1W, fc1b, T1, N_GRAPHS, 512, 128);
  k_fc<<<blocks(N_GRAPHS * 64),  256, 0, stream>>>(T1, fc2W, fc2b, T2, N_GRAPHS, 128, 64);
  k_fc<<<blocks(N_GRAPHS * 10),  256, 0, stream>>>(T2, fc3W, fc3b, (float*)d_out, N_GRAPHS, 64, 10);
}
